// GVP_GNN_38757784879462
// MI455X (gfx1250) — matrix operand order repaired and checked by reading, not compiled, not run on hardware
//
#include <hip/hip_runtime.h>
#include <hip/hip_fp16.h>

// ---------------------------------------------------------------------------
// GVP-GNN forward for MI455X (gfx1250, wave32).
// 16 rows (edges/nodes) per 128-thread WG; activations staged in LDS as f16;
// every linear runs on v_wmma_f32_16x16x32_f16 (f32 accumulate).
// Weights pre-transposed/padded/converted to f16 once per call: B fragment =
// 2x global_load_b128, A fragment = 2x ds_load_b128, no bounds logic in the
// GEMM inner loop. Staging loops avoid integer div/mod entirely; the edge
// gather is float4-vectorized.
// ---------------------------------------------------------------------------

#define NNODES 20000
#define NEDGES 320000
#define NGRAPH 64
#define NSd 100
#define NVd 16

typedef __attribute__((ext_vector_type(16))) _Float16 v16h;
typedef __attribute__((ext_vector_type(8)))  _Float16 v8h;
typedef __attribute__((ext_vector_type(4)))  _Float16 v4h;
typedef __attribute__((ext_vector_type(8)))  float    v8f;
typedef __attribute__((ext_vector_type(4)))  float    v4f;

typedef __attribute__((address_space(1))) const float    g_cfloat;
typedef __attribute__((address_space(1))) const int      g_cint;
typedef __attribute__((address_space(1))) float          g_float;
typedef __attribute__((address_space(1))) const v8h      g_cv8h;
typedef __attribute__((address_space(1))) const v4f      g_cv4f;
typedef __attribute__((address_space(3))) const v8h      l_cv8h;
typedef __attribute__((address_space(3))) float          l_float;
typedef __attribute__((address_space(3))) _Float16       l_half;
typedef __attribute__((address_space(3))) v4h            l_v4h;

// raw f32 params as passed in d_in
struct GvpP { const float *wh, *ws_w, *ws_b, *wv, *wsv_w, *wsv_b; };
// prepped params: f16 transposed padded weights + padded f32 biases
struct GvpW { const _Float16 *wh, *ws, *wv, *wsv; const float *bs, *bsv; };

__device__ __forceinline__ int imin(int a, int b) { return a < b ? a : b; }
__device__ __forceinline__ float sigmoidf(float x) { return 1.f / (1.f + __expf(-x)); }
__device__ __forceinline__ float ldg(const float* p) { return *(g_cfloat*)p; }
__device__ __forceinline__ int   ldgi(const int* p)  { return *(g_cint*)p; }
__device__ __forceinline__ void  stg(float* p, float x) { *(g_float*)p = x; }
__device__ __forceinline__ v4f   ldg4(const float* p) { return *(g_cv4f*)p; }
__device__ __forceinline__ void  st4h(_Float16* p, v4f x) {
  v4h h;
  h[0] = (_Float16)x[0]; h[1] = (_Float16)x[1];
  h[2] = (_Float16)x[2]; h[3] = (_Float16)x[3];
  *(l_v4h*)((l_half*)p) = h;
}

// ----- WMMA fragment helpers (layouts per CDNA5 ISA) -------------

// A fragment 16x32 f16 from LDS: lane holds row M=lane&15, K {0..7,16..23}
// (+8 for upper lane half) => two contiguous 8-half chunks => 2x ds_load_b128.
__device__ __forceinline__ v16h frag_a(const _Float16* A, int lda, int k0) {
  int lane = threadIdx.x & 31;
  int m = lane & 15;
  int kh = (lane >> 4) << 3;
  const l_cv8h* p = (const l_cv8h*)((const l_half*)A + m * lda + k0 + kh);
  v8h lo = p[0];   // K +0..7
  v8h hi = p[2];   // K +16..23
  return __builtin_shufflevector(lo, hi, 0,1,2,3,4,5,6,7,8,9,10,11,12,13,14,15);
}

// B fragment 32x16 f16 from prepped transposed weights WT[NP][KP]:
// lane holds col N=n0+(lane&15), K {0..7,16..23} (+8 for upper lane half) => two global_load_b128.
__device__ __forceinline__ v16h frag_bt(const _Float16* WT, int KP, int k0, int n0) {
  int lane = threadIdx.x & 31;
  int n = n0 + (lane & 15);
  int kb = k0 + ((lane >> 4) << 3);
  const g_cv8h* p = (const g_cv8h*)((const __attribute__((address_space(1))) _Float16*)WT
                                    + n * KP + kb);
  v8h lo = p[0];
  v8h hi = p[2];
  return __builtin_shufflevector(lo, hi, 0,1,2,3,4,5,6,7,8,9,10,11,12,13,14,15);
}

__device__ __forceinline__ void wmma_store(float* OUT, int ldo, v8f c, int n0,
                                           const float* biasP) {
  int lane = threadIdx.x & 31;
  int n = n0 + (lane & 15);
  float bb = *((g_cfloat*)biasP + n);
  int mb = (lane >> 4) << 3;
  l_float* o = (l_float*)OUT;
#pragma unroll
  for (int r = 0; r < 8; ++r) o[(mb + r) * ldo + n] = c[r] + bb;
}

__device__ __forceinline__ void wmma_store_nb(float* OUT, int ldo, v8f c, int n0) {
  int lane = threadIdx.x & 31;
  int n = n0 + (lane & 15);
  int mb = (lane >> 4) << 3;
  l_float* o = (l_float*)OUT;
#pragma unroll
  for (int r = 0; r < 8; ++r) o[(mb + r) * ldo + n] = c[r];
}

__device__ __forceinline__ void wmma_store_h(_Float16* OUT, int ldo, v8f c, int n0,
                                             const float* biasP) {
  int lane = threadIdx.x & 31;
  int n = n0 + (lane & 15);
  float bb = *((g_cfloat*)biasP + n);
  int mb = (lane >> 4) << 3;
  l_half* o = (l_half*)OUT;
#pragma unroll
  for (int r = 0; r < 8; ++r) o[(mb + r) * ldo + n] = (_Float16)(c[r] + bb);
}

// Workgroup GEMM: OUT[16][NP] = A[16][KP](f16 LDS) @ WT + bias (all padded).
// 4 waves split N tiles; EXEC stays full around the WMMAs; K loop unrolled.
template <int KP>
__device__ void wg_gemm(const _Float16* A, const _Float16* WT, const float* biasP,
                        int NP, float* OUT, int ldo) {
  int wave = threadIdx.x >> 5;
  int nt = NP >> 4;
  for (int t = wave; t < nt; t += 4) {
    int n0 = t << 4;
    v8f c = {0.f, 0.f, 0.f, 0.f, 0.f, 0.f, 0.f, 0.f};
#pragma unroll
    for (int k0 = 0; k0 < KP; k0 += 32) {
      v16h a = frag_a(A, KP, k0);
      v16h b = frag_bt(WT, KP, k0, n0);
      c = __builtin_amdgcn_wmma_f32_16x16x32_f16(false, a, false, b, (short)0, c, false, false);
    }
    wmma_store(OUT, ldo, c, n0, biasP);
  }
}

template <int KP>
__device__ void wg_gemm_h(const _Float16* A, const _Float16* WT, const float* biasP,
                          int NP, _Float16* OUT, int ldo) {
  int wave = threadIdx.x >> 5;
  int nt = NP >> 4;
  for (int t = wave; t < nt; t += 4) {
    int n0 = t << 4;
    v8f c = {0.f, 0.f, 0.f, 0.f, 0.f, 0.f, 0.f, 0.f};
#pragma unroll
    for (int k0 = 0; k0 < KP; k0 += 32) {
      v16h a = frag_a(A, KP, k0);
      v16h b = frag_bt(WT, KP, k0, n0);
      c = __builtin_amdgcn_wmma_f32_16x16x32_f16(false, a, false, b, (short)0, c, false, false);
    }
    wmma_store_h(OUT, ldo, c, n0, biasP);
  }
}

// Same GEMM applied to x/y/z coordinate planes (shared WT), no bias.
template <int KP>
__device__ void wg_gemm_xyz(const _Float16* A, int as, const _Float16* WT,
                            int NP, float* OUT, int ldo, int os) {
  int wave = threadIdx.x >> 5;
  int nt = NP >> 4;
  for (int t = wave; t < 3 * nt; t += 4) {
    int c3 = t % 3, tt = t / 3;
    int n0 = tt << 4;
    v8f c = {0.f, 0.f, 0.f, 0.f, 0.f, 0.f, 0.f, 0.f};
#pragma unroll
    for (int k0 = 0; k0 < KP; k0 += 32) {
      v16h a = frag_a(A + c3 * as, KP, k0);
      v16h b = frag_bt(WT, KP, k0, n0);
      c = __builtin_amdgcn_wmma_f32_16x16x32_f16(false, a, false, b, (short)0, c, false, false);
    }
    wmma_store_nb(OUT + c3 * os, ldo, c, n0);
  }
}

// ----- fused GVP (si=100, vi=16 -> so=100, vo=16, vector_gate) -------------
__device__ void gvp_small(const GvpW g, bool relu_s, bool sig_gate,
                          _Float16* sA, _Float16* vA, float* scr,
                          float* sOut, _Float16* gA, float* gOut) {
  const int tid = threadIdx.x;
  wg_gemm_xyz<32>(vA, 512, g.wh, 16, scr, 16, 256);         // vh
  __syncthreads();
  for (int idx = tid; idx < 256; idx += 128) {
    int r = idx >> 4, h = idx & 15;
    float x = scr[idx], y = scr[256 + idx], z = scr[512 + idx];
    sA[r * 128 + 100 + h] = (_Float16)sqrtf(fmaxf(x * x + y * y + z * z, 1e-8f));
    vA[r * 32 + h] = (_Float16)x;
    vA[512 + r * 32 + h] = (_Float16)y;
    vA[1024 + r * 32 + h] = (_Float16)z;
  }
  __syncthreads();
  wg_gemm<128>(sA, g.ws, g.bs, 112, sOut, 112);             // s = [s,vn] @ ws
  wg_gemm_xyz<32>(vA, 512, g.wv, 16, scr, 16, 256);         // vo = vh @ wv
  __syncthreads();
  for (int idx = tid; idx < 16 * 128; idx += 128) {
    int r = idx >> 7, k = idx & 127;
    float val = 0.f;
    if (k < 100) { float x = sOut[r * 112 + k]; val = sig_gate ? sigmoidf(x) : x; }
    gA[idx] = (_Float16)val;
  }
  __syncthreads();
  wg_gemm<128>(gA, g.wsv, g.bsv, 16, gOut, 16);             // gate logits
  __syncthreads();
  for (int idx = tid; idx < 512; idx += 128) {
    int r = idx >> 5, k = idx & 31;
    float gg = (k < 16) ? sigmoidf(gOut[r * 16 + k]) : 0.f;
    vA[idx]        = (_Float16)((k < 16) ? scr[r * 16 + k] * gg : 0.f);
    vA[512 + idx]  = (_Float16)((k < 16) ? scr[256 + r * 16 + k] * gg : 0.f);
    vA[1024 + idx] = (_Float16)((k < 16) ? scr[512 + r * 16 + k] * gg : 0.f);
  }
  if (relu_s) {
    for (int idx = tid; idx < 16 * 128; idx += 128) {
      int r = idx >> 7, k = idx & 127;
      float val = (k < 100) ? fmaxf(sOut[r * 112 + k], 0.f) : 0.f;
      sA[idx] = (_Float16)val;
    }
  }
  __syncthreads();
}

// ---------------------------------------------------------------------------
// Edge message kernel: 16 edges per workgroup, 3 chained GVPs, atomic scatter.
// ---------------------------------------------------------------------------
__launch_bounds__(128)
__global__ void edge_msg_kernel(const float* __restrict__ s, const float* __restrict__ v,
                                const float* __restrict__ es, const float* __restrict__ ev,
                                const int* __restrict__ ei,
                                GvpW g0, GvpW g1, GvpW g2,
                                float* __restrict__ dsum, float* __restrict__ dvsum) {
  __shared__ alignas(16) _Float16 sA[16 * 288];
  __shared__ alignas(16) _Float16 vA[3 * 16 * 64];
  __shared__ alignas(16) float    vhF[3 * 16 * 48];
  __shared__ alignas(16) float    voF[3 * 16 * 16];
  __shared__ alignas(16) float    sOut[16 * 112];
  __shared__ alignas(16) _Float16 gA[16 * 128];
  __shared__ alignas(16) float    gOut[16 * 16];
  __shared__ int srcI[16], dstI[16];

  const int tid = threadIdx.x;
  const int lane = tid & 31;
  const int wv = tid >> 5;
  const int e0 = blockIdx.x << 4;

  if (tid < 16) {
    int ec = imin(e0 + tid, NEDGES - 1);
    srcI[tid] = ldgi(ei + ec);
    dstI[tid] = ldgi(ei + NEDGES + ec);
  }
  __syncthreads();

  // gather (float4): ms = [s_src(100), es(32), s_dst(100)] pad->288.
  // 4-float chunks: 0..24 s_src | 25..32 es | 33..57 s_dst | 58..71 zero.
#pragma unroll
  for (int rr = 0; rr < 4; ++rr) {
    int r = wv * 4 + rr;
    int ec = imin(e0 + r, NEDGES - 1);
    int si = srcI[r], di = dstI[r];
    for (int k4 = lane; k4 < 72; k4 += 32) {
      v4f x = {0.f, 0.f, 0.f, 0.f};
      if (k4 < 25) x = ldg4(s + si * NSd + k4 * 4);
      else if (k4 < 33) x = ldg4(es + ec * 32 + (k4 - 25) * 4);
      else if (k4 < 58) x = ldg4(s + di * NSd + (k4 - 33) * 4);
      st4h(sA + r * 288 + k4 * 4, x);
    }
  }
  // gather: mv = [v_src(16), ev(1), v_dst(16)] per coord, K pad->64
  for (int idx = tid; idx < 3 * 16 * 64; idx += 128) {
    int c = idx >> 10, rem = idx & 1023;
    int r = rem >> 6, k = rem & 63;
    int ec = imin(e0 + r, NEDGES - 1);
    float val = 0.f;
    if (k < 16) val = ldg(v + (srcI[r] * NVd + k) * 3 + c);
    else if (k == 16) val = ldg(ev + ec * 3 + c);
    else if (k < 33) val = ldg(v + (dstI[r] * NVd + (k - 17)) * 3 + c);
    vA[idx] = (_Float16)val;
  }
  __syncthreads();

  // ---- GVP0: si=232, vi=33, h=33, so=100, vo=16 ----
  wg_gemm_xyz<64>(vA, 1024, g0.wh, 48, vhF, 48, 768);
  __syncthreads();
  for (int idx = tid; idx < 1024; idx += 128) {
    int r = idx >> 6, h = idx & 63;
    if (h < 33) {
      float x = vhF[r * 48 + h], y = vhF[768 + r * 48 + h], z = vhF[1536 + r * 48 + h];
      sA[r * 288 + 232 + h] = (_Float16)sqrtf(fmaxf(x * x + y * y + z * z, 1e-8f));
      vA[r * 64 + h] = (_Float16)x;
      vA[1024 + r * 64 + h] = (_Float16)y;
      vA[2048 + r * 64 + h] = (_Float16)z;
    }
  }
  __syncthreads();
  wg_gemm<288>(sA, g0.ws, g0.bs, 112, sOut, 112);
  wg_gemm_xyz<64>(vA, 1024, g0.wv, 16, voF, 16, 256);
  __syncthreads();
  for (int idx = tid; idx < 16 * 128; idx += 128) {
    int r = idx >> 7, k = idx & 127;
    float val = (k < 100) ? sigmoidf(sOut[r * 112 + k]) : 0.f;
    gA[idx] = (_Float16)val;
  }
  __syncthreads();
  wg_gemm<128>(gA, g0.wsv, g0.bsv, 16, gOut, 16);
  __syncthreads();
  for (int idx = tid; idx < 512; idx += 128) {
    int r = idx >> 5, k = idx & 31;
    float gg = (k < 16) ? sigmoidf(gOut[r * 16 + k]) : 0.f;
    vA[idx]        = (_Float16)((k < 16) ? voF[r * 16 + k] * gg : 0.f);
    vA[512 + idx]  = (_Float16)((k < 16) ? voF[256 + r * 16 + k] * gg : 0.f);
    vA[1024 + idx] = (_Float16)((k < 16) ? voF[512 + r * 16 + k] * gg : 0.f);
  }
  for (int idx = tid; idx < 16 * 128; idx += 128) {
    int r = idx >> 7, k = idx & 127;
    float val = (k < 100) ? fmaxf(sOut[r * 112 + k], 0.f) : 0.f;
    sA[r * 128 + k] = (_Float16)val;          // re-layout: stride 128 now
  }
  __syncthreads();

  gvp_small(g1, true, true, sA, vA, vhF, sOut, gA, gOut);
  gvp_small(g2, false, false, sA, vA, vhF, sOut, gA, gOut);

  // scatter messages to destination nodes (mean done in node kernel)
  for (int idx = tid; idx < 16 * 128; idx += 128) {
    int r = idx >> 7, k = idx & 127;
    if (k < 100 && e0 + r < NEDGES)
      atomicAdd(&dsum[dstI[r] * NSd + k], sOut[r * 112 + k]);
  }
  for (int idx = tid; idx < 16 * 64; idx += 128) {
    int r = idx >> 6, q = idx & 63;
    if (q < 48 && e0 + r < NEDGES) {
      int vi = q / 3, c = q - vi * 3;
      atomicAdd(&dvsum[dstI[r] * 48 + q], (float)vA[c * 512 + r * 32 + vi]);
    }
  }
}

// ---------------------------------------------------------------------------
// Node update: residual + LN + feed-forward GVPs + LN. 16 nodes per WG.
// ---------------------------------------------------------------------------
__launch_bounds__(128)
__global__ void node_update_kernel(float* __restrict__ s, float* __restrict__ v,
                                   const float* __restrict__ dsum,
                                   const float* __restrict__ dvsum,
                                   const float* __restrict__ cnt,
                                   GvpW f0, GvpW f1,
                                   const float* __restrict__ ln0g, const float* __restrict__ ln0b,
                                   const float* __restrict__ ln1g, const float* __restrict__ ln1b) {
  __shared__ alignas(16) _Float16 sA[16 * 448];    // ff0 A(160)/gate A(416)/ff1 A(448)
  __shared__ alignas(16) _Float16 sOutH[16 * 400]; // ff0 s output (f16)
  __shared__ alignas(16) float    sOut[16 * 112];  // ff1 s output
  __shared__ alignas(16) _Float16 vA[3 * 16 * 32];
  __shared__ alignas(16) float    vhF[3 * 16 * 32];
  __shared__ alignas(16) float    voF[3 * 16 * 32];
  __shared__ alignas(16) float    sR[16 * 100];
  __shared__ alignas(16) float    vR[16 * 48];
  __shared__ alignas(16) float    gOut[16 * 32];
  __shared__ float invc[16], rowa[16], rowb[16];

  const int tid = threadIdx.x;
  const int n0 = blockIdx.x << 4;

  if (tid < 16) invc[tid] = 1.f / fmaxf(ldg(cnt + imin(n0 + tid, NNODES - 1)), 1.f);
  __syncthreads();

  for (int idx = tid; idx < 16 * 128; idx += 128) {
    int r = idx >> 7, k = idx & 127;
    if (k < 100) {
      int n = imin(n0 + r, NNODES - 1);
      sR[r * 100 + k] = ldg(s + n * NSd + k) + ldg(dsum + n * NSd + k) * invc[r];
    }
  }
  for (int idx = tid; idx < 16 * 64; idx += 128) {
    int r = idx >> 6, q = idx & 63;
    if (q < 48) {
      int n = imin(n0 + r, NNODES - 1);
      vR[r * 48 + q] = ldg(v + n * 48 + q) + ldg(dvsum + n * 48 + q) * invc[r];
    }
  }
  __syncthreads();
  // LN0 stats (wave0) and vector-norm scale (wave1, reuse invc)
  if (tid < 16) {
    float mu = 0.f;
    for (int k = 0; k < 100; ++k) mu += sR[tid * 100 + k];
    mu *= 0.01f;
    float var = 0.f;
    for (int k = 0; k < 100; ++k) { float d = sR[tid * 100 + k] - mu; var += d * d; }
    rowa[tid] = mu;
    rowb[tid] = rsqrtf(var * 0.01f + 1e-5f);
  } else if (tid >= 32 && tid < 48) {
    int r = tid - 32;
    float m = 0.f;
    for (int vi = 0; vi < 16; ++vi) {
      float n2 = 0.f;
      for (int c = 0; c < 3; ++c) { float x = vR[r * 48 + vi * 3 + c]; n2 += x * x; }
      m += fmaxf(n2, 1e-8f);
    }
    invc[r] = rsqrtf(m * (1.f / 16.f));
  }
  __syncthreads();
  // apply LN0; build ff0 A (stride 160): cols 0..99 s, 100..131 vn (later)
#pragma unroll 1
  for (int r = 0; r < 16; ++r)
    for (int k = tid; k < 160; k += 128) {
      float val = 0.f;
      if (k < 100) {
        float x = (sR[r * 100 + k] - rowa[r]) * rowb[r] * ldg(ln0g + k) + ldg(ln0b + k);
        sR[r * 100 + k] = x;
        val = x;
      }
      sA[r * 160 + k] = (_Float16)val;
    }
  for (int idx = tid; idx < 16 * 64; idx += 128) {
    int r = idx >> 6, q = idx & 63;
    if (q < 48) vR[r * 48 + q] *= invc[r];
  }
  __syncthreads();
  for (int idx = tid; idx < 512; idx += 128) {
    int r = idx >> 5, k = idx & 31;
    vA[idx]        = (_Float16)((k < 16) ? vR[r * 48 + k * 3 + 0] : 0.f);
    vA[512 + idx]  = (_Float16)((k < 16) ? vR[r * 48 + k * 3 + 1] : 0.f);
    vA[1024 + idx] = (_Float16)((k < 16) ? vR[r * 48 + k * 3 + 2] : 0.f);
  }
  __syncthreads();

  // ---- FF0: 100/16 -> 400/32 ----
  wg_gemm_xyz<32>(vA, 512, f0.wh, 32, vhF, 32, 512);
  __syncthreads();
  for (int idx = tid; idx < 512; idx += 128) {
    int r = idx >> 5, h = idx & 31;
    float x = vhF[idx], y = vhF[512 + idx], z = vhF[1024 + idx];
    sA[r * 160 + 100 + h] = (_Float16)sqrtf(fmaxf(x * x + y * y + z * z, 1e-8f));
    vA[idx] = (_Float16)x; vA[512 + idx] = (_Float16)y; vA[1024 + idx] = (_Float16)z;
  }
  __syncthreads();
  wg_gemm_h<160>(sA, f0.ws, f0.bs, 400, sOutH, 400);
  wg_gemm_xyz<32>(vA, 512, f0.wv, 32, voF, 32, 512);
  __syncthreads();
  // gate A (stride 416) aliases sA (ff0 A no longer needed)
#pragma unroll 1
  for (int r = 0; r < 16; ++r)
    for (int k = tid; k < 416; k += 128) {
      float val = (k < 400) ? sigmoidf((float)sOutH[r * 400 + k]) : 0.f;
      sA[r * 416 + k] = (_Float16)val;
    }
  __syncthreads();
  wg_gemm<416>(sA, f0.wsv, f0.bsv, 32, gOut, 32);
  __syncthreads();
  for (int idx = tid; idx < 512; idx += 128) {
    int r = idx >> 5, k = idx & 31;
    float gg = sigmoidf(gOut[r * 32 + k]);
    vA[idx]        = (_Float16)(voF[idx] * gg);
    vA[512 + idx]  = (_Float16)(voF[512 + idx] * gg);
    vA[1024 + idx] = (_Float16)(voF[1024 + idx] * gg);
  }
#pragma unroll 1
  for (int r = 0; r < 16; ++r)
    for (int k = tid; k < 448; k += 128) {
      float val = (k < 400) ? fmaxf((float)sOutH[r * 400 + k], 0.f) : 0.f;
      sA[r * 448 + k] = (_Float16)val;       // ff1 A, stride 448
    }
  __syncthreads();

  // ---- FF1: 400/32 -> 100/16 ----
  wg_gemm_xyz<32>(vA, 512, f1.wh, 32, vhF, 32, 512);
  __syncthreads();
  for (int idx = tid; idx < 512; idx += 128) {
    int r = idx >> 5, h = idx & 31;
    float x = vhF[idx], y = vhF[512 + idx], z = vhF[1024 + idx];
    sA[r * 448 + 400 + h] = (_Float16)sqrtf(fmaxf(x * x + y * y + z * z, 1e-8f));
    vA[idx] = (_Float16)x; vA[512 + idx] = (_Float16)y; vA[1024 + idx] = (_Float16)z;
  }
  __syncthreads();
  wg_gemm<448>(sA, f1.ws, f1.bs, 112, sOut, 112);
  wg_gemm_xyz<32>(vA, 512, f1.wv, 16, voF, 16, 256);
  __syncthreads();
  for (int idx = tid; idx < 16 * 128; idx += 128) {
    int r = idx >> 7, k = idx & 127;
    float val = (k < 100) ? sOut[r * 112 + k] : 0.f;  // vec act = identity here
    sA[idx] = (_Float16)val;                          // reuse as gate A (128)
  }
  __syncthreads();
  wg_gemm<128>(sA, f1.wsv, f1.bsv, 16, gOut, 16);
  __syncthreads();

  // residual + LN1
  for (int idx = tid; idx < 16 * 128; idx += 128) {
    int r = idx >> 7, k = idx & 127;
    if (k < 100) sR[r * 100 + k] += sOut[r * 112 + k];
  }
  for (int idx = tid; idx < 16 * 64; idx += 128) {
    int r = idx >> 6, q = idx & 63;
    if (q < 48) {
      int vi = q / 3, c = q - vi * 3;
      vR[r * 48 + q] += voF[c * 256 + r * 16 + vi] * sigmoidf(gOut[r * 16 + vi]);
    }
  }
  __syncthreads();
  if (tid < 16) {
    float mu = 0.f;
    for (int k = 0; k < 100; ++k) mu += sR[tid * 100 + k];
    mu *= 0.01f;
    float var = 0.f;
    for (int k = 0; k < 100; ++k) { float d = sR[tid * 100 + k] - mu; var += d * d; }
    rowa[tid] = mu;
    rowb[tid] = rsqrtf(var * 0.01f + 1e-5f);
  } else if (tid >= 32 && tid < 48) {
    int r = tid - 32;
    float m = 0.f;
    for (int vi = 0; vi < 16; ++vi) {
      float n2 = 0.f;
      for (int c = 0; c < 3; ++c) { float x = vR[r * 48 + vi * 3 + c]; n2 += x * x; }
      m += fmaxf(n2, 1e-8f);
    }
    invc[r] = rsqrtf(m * (1.f / 16.f));
  }
  __syncthreads();
  for (int idx = tid; idx < 16 * 128; idx += 128) {
    int r = idx >> 7, k = idx & 127;
    int n = n0 + r;
    if (k < 100 && n < NNODES)
      stg(s + n * NSd + k, (sR[r * 100 + k] - rowa[r]) * rowb[r] * ldg(ln1g + k) + ldg(ln1b + k));
  }
  for (int idx = tid; idx < 16 * 64; idx += 128) {
    int r = idx >> 6, q = idx & 63;
    int n = n0 + r;
    if (q < 48 && n < NNODES) stg(v + n * 48 + q, vR[r * 48 + q] * invc[r]);
  }
}

// ---------------------------------------------------------------------------
// Input embeddings
// ---------------------------------------------------------------------------
__launch_bounds__(128)
__global__ void embed_nodes_kernel(const float* __restrict__ hs, const float* __restrict__ hv,
                                   GvpW g, float* __restrict__ s, float* __restrict__ v) {
  __shared__ alignas(16) _Float16 sA[16 * 32];
  __shared__ alignas(16) _Float16 vA[3 * 16 * 32];
  __shared__ alignas(16) float    vhF[3 * 16 * 16];
  __shared__ alignas(16) float    sOut[16 * 112];
  __shared__ alignas(16) float    voF[3 * 16 * 16];
  const int tid = threadIdx.x;
  const int n0 = blockIdx.x << 4;
  for (int idx = tid; idx < 512; idx += 128) {
    int r = idx >> 5, k = idx & 31;
    int n = imin(n0 + r, NNODES - 1);
    vA[idx]        = (_Float16)((k < 3) ? ldg(hv + (n * 3 + k) * 3 + 0) : 0.f);
    vA[512 + idx]  = (_Float16)((k < 3) ? ldg(hv + (n * 3 + k) * 3 + 1) : 0.f);
    vA[1024 + idx] = (_Float16)((k < 3) ? ldg(hv + (n * 3 + k) * 3 + 2) : 0.f);
  }
  __syncthreads();
  wg_gemm_xyz<32>(vA, 512, g.wh, 16, vhF, 16, 256);
  __syncthreads();
  for (int idx = tid; idx < 512; idx += 128) {
    int r = idx >> 5, k = idx & 31;
    int n = imin(n0 + r, NNODES - 1);
    float val = 0.f;
    if (k < 6) val = ldg(hs + n * 6 + k);
    else if (k < 22) {
      int h = k - 6;
      float x = vhF[r * 16 + h], y = vhF[256 + r * 16 + h], z = vhF[512 + r * 16 + h];
      val = sqrtf(fmaxf(x * x + y * y + z * z, 1e-8f));
    }
    sA[idx] = (_Float16)val;
  }
  for (int idx = tid; idx < 512; idx += 128) {
    int r = idx >> 5, k = idx & 31;
    vA[idx]        = (_Float16)((k < 16) ? vhF[r * 16 + k] : 0.f);
    vA[512 + idx]  = (_Float16)((k < 16) ? vhF[256 + r * 16 + k] : 0.f);
    vA[1024 + idx] = (_Float16)((k < 16) ? vhF[512 + r * 16 + k] : 0.f);
  }
  __syncthreads();
  wg_gemm<32>(sA, g.ws, g.bs, 112, sOut, 112);
  wg_gemm_xyz<32>(vA, 512, g.wv, 16, voF, 16, 256);
  __syncthreads();
  for (int idx = tid; idx < 16 * 128; idx += 128) {
    int r = idx >> 7, k = idx & 127;
    int n = n0 + r;
    if (k < 100 && n < NNODES) stg(s + n * NSd + k, sOut[r * 112 + k]);
  }
  for (int idx = tid; idx < 16 * 64; idx += 128) {
    int r = idx >> 6, q = idx & 63;
    int n = n0 + r;
    if (q < 48 && n < NNODES) {
      int vi = q / 3, c = q - vi * 3;
      stg(v + n * 48 + q, voF[c * 256 + r * 16 + vi]);
    }
  }
}

__launch_bounds__(128)
__global__ void embed_edges_kernel(const float* __restrict__ hes, const float* __restrict__ hev,
                                   GvpW g, const float* __restrict__ whr,
                                   const float* __restrict__ wvr,
                                   float* __restrict__ es, float* __restrict__ ev) {
  __shared__ alignas(16) _Float16 sA[16 * 64];
  __shared__ alignas(16) float    sOut[16 * 32];
  __shared__ alignas(16) float    vh[16 * 3];
  const int tid = threadIdx.x;
  const int e0 = blockIdx.x << 4;
  const float wh00 = ldg(whr);
  if (tid < 48) {
    int r = tid / 3, c = tid - r * 3;
    int e = imin(e0 + r, NEDGES - 1);
    vh[r * 3 + c] = ldg(hev + e * 3 + c) * wh00;
  }
  __syncthreads();
  for (int idx = tid; idx < 16 * 64; idx += 128) {
    int r = idx >> 6, k = idx & 63;
    int e = imin(e0 + r, NEDGES - 1);
    float val = 0.f;
    if (k < 32) val = ldg(hes + e * 32 + k);
    else if (k == 32) {
      float x = vh[r * 3], y = vh[r * 3 + 1], z = vh[r * 3 + 2];
      val = sqrtf(fmaxf(x * x + y * y + z * z, 1e-8f));
    }
    sA[idx] = (_Float16)val;
  }
  __syncthreads();
  wg_gemm<64>(sA, g.ws, g.bs, 32, sOut, 32);
  __syncthreads();
  const float wv00 = ldg(wvr);
  for (int idx = tid; idx < 16 * 32; idx += 128) {
    int r = idx >> 5;
    int e = e0 + r;
    if (e < NEDGES) stg(es + e * 32 + (idx & 31), sOut[idx]);
  }
  if (tid < 48) {
    int r = tid / 3;
    int e = e0 + r;
    if (e < NEDGES) stg(ev + e * 3 + (tid - r * 3), vh[tid] * wv00);
  }
}

// ---------------------------------------------------------------------------
// Output head: W_out GVP (relu) + graph mean pooling via atomics
// ---------------------------------------------------------------------------
__launch_bounds__(128)
__global__ void out_pool_kernel(const float* __restrict__ s, const float* __restrict__ v,
                                const int* __restrict__ batch, GvpW g,
                                float* __restrict__ pooled) {
  __shared__ alignas(16) _Float16 sA[16 * 128];
  __shared__ alignas(16) _Float16 vA[3 * 16 * 32];
  __shared__ alignas(16) float    vhF[3 * 16 * 16];
  __shared__ alignas(16) float    sOut[16 * 112];
  const int tid = threadIdx.x;
  const int n0 = blockIdx.x << 4;
  for (int idx = tid; idx < 512; idx += 128) {
    int r = idx >> 5, k = idx & 31;
    int n = imin(n0 + r, NNODES - 1);
    vA[idx]        = (_Float16)((k < 16) ? ldg(v + n * 48 + k * 3 + 0) : 0.f);
    vA[512 + idx]  = (_Float16)((k < 16) ? ldg(v + n * 48 + k * 3 + 1) : 0.f);
    vA[1024 + idx] = (_Float16)((k < 16) ? ldg(v + n * 48 + k * 3 + 2) : 0.f);
  }
  __syncthreads();
  wg_gemm_xyz<32>(vA, 512, g.wh, 16, vhF, 16, 256);
  __syncthreads();
  for (int idx = tid; idx < 16 * 128; idx += 128) {
    int r = idx >> 7, k = idx & 127;
    int n = imin(n0 + r, NNODES - 1);
    float val = 0.f;
    if (k < 100) val = ldg(s + n * NSd + k);
    else if (k < 116) {
      int h = k - 100;
      float x = vhF[r * 16 + h], y = vhF[256 + r * 16 + h], z = vhF[512 + r * 16 + h];
      val = sqrtf(fmaxf(x * x + y * y + z * z, 1e-8f));
    }
    sA[idx] = (_Float16)val;
  }
  __syncthreads();
  wg_gemm<128>(sA, g.ws, g.bs, 112, sOut, 112);
  __syncthreads();
  for (int idx = tid; idx < 16 * 128; idx += 128) {
    int r = idx >> 7, k = idx & 127;
    int n = n0 + r;
    if (k < 100 && n < NNODES)
      atomicAdd(&pooled[ldgi(batch + n) * NSd + k], fmaxf(sOut[r * 112 + k], 0.f));
  }
}

__launch_bounds__(64)
__global__ void dense_kernel(const float* __restrict__ pooled, const float* __restrict__ gcnt,
                             const float* __restrict__ w1, const float* __restrict__ b1,
                             const float* __restrict__ w2, const float* __restrict__ b2,
                             float* __restrict__ out) {
  int g = threadIdx.x;
  if (g >= NGRAPH) return;
  float ic = 1.f / fmaxf(ldg(gcnt + g), 1.f);
  float acc = ldg(b2);
  for (int j = 0; j < NSd; ++j) {
    float h = ldg(b1 + j);
    for (int k = 0; k < NSd; ++k) h += ldg(pooled + g * NSd + k) * ic * ldg(w1 + k * NSd + j);
    acc += (h * sigmoidf(h)) * ldg(w2 + j);
  }
  stg(out + g, acc);
}

// ---------------------------------------------------------------------------
// Utility / prep kernels
// ---------------------------------------------------------------------------
__global__ void zero_kernel(float* p, int n) {
  int i = blockIdx.x * blockDim.x + threadIdx.x;
  if (i < n) stg(p + i, 0.f);
}
__global__ void degree_kernel(const int* __restrict__ ei, float* __restrict__ cnt) {
  int e = blockIdx.x * blockDim.x + threadIdx.x;
  if (e < NEDGES) atomicAdd(&cnt[ldgi(ei + NEDGES + e)], 1.f);
}
__global__ void gcount_kernel(const int* __restrict__ batch, float* __restrict__ gcnt) {
  int n = blockIdx.x * blockDim.x + threadIdx.x;
  if (n < NNODES) atomicAdd(&gcnt[ldgi(batch + n)], 1.f);
}
// Transpose + pad + cvt weight W[K][N] f32 -> WT[NP][KP] f16 (zeros in pad).
__global__ void prep_wt_kernel(const float* __restrict__ W, _Float16* __restrict__ WT,
                               int K, int N, int KP, int NP) {
  int i = blockIdx.x * blockDim.x + threadIdx.x;
  if (i >= NP * KP) return;
  int n = i / KP, k = i - n * KP;
  float val = (n < N && k < K) ? ldg(W + k * N + n) : 0.f;
  *((__attribute__((address_space(1))) _Float16*)WT + i) = (_Float16)val;
}
__global__ void prep_bias_kernel(const float* __restrict__ b, float* __restrict__ bp,
                                 int N, int NP) {
  int i = blockIdx.x * blockDim.x + threadIdx.x;
  if (i < NP) stg(bp + i, (i < N) ? ldg(b + i) : 0.f);
}

// ---------------------------------------------------------------------------
// Launcher. Param leaves assumed flattened in reference dict insertion order
// (None leaves skipped): W_v(4), W_e(4), 5 x [msg0(6),msg1(6),msg2(6),
// ff0(6), ff1(6), ln0(2), ln1(2)], W_out(3), dense(4).
// ---------------------------------------------------------------------------
extern "C" void kernel_launch(void* const* d_in, const int* in_sizes, int n_in,
                              void* d_out, int out_size, void* d_ws, size_t ws_size,
                              hipStream_t stream) {
  const float* hVs = (const float*)d_in[0];
  const float* hVv = (const float*)d_in[1];
  const float* hEs = (const float*)d_in[2];
  const float* hEv = (const float*)d_in[3];
  const int*   ei  = (const int*)d_in[4];
  const int*   bat = (const int*)d_in[5];
  // d_in[6] pembedding: unused by the reference

  int p = 7;
  auto P = [&]() { return (const float*)d_in[p++]; };
  auto GV = [&](bool gate) {
    GvpP g;
    g.wh = P(); g.ws_w = P(); g.ws_b = P(); g.wv = P();
    if (gate) { g.wsv_w = P(); g.wsv_b = P(); } else { g.wsv_w = nullptr; g.wsv_b = nullptr; }
    return g;
  };
  GvpP rWv = GV(false);
  GvpP rWe = GV(false);
  struct LayerP { GvpP m0, m1, m2, f0, f1; const float *l0g, *l0b, *l1g, *l1b; } L[5];
  for (int i = 0; i < 5; ++i) {
    L[i].m0 = GV(true); L[i].m1 = GV(true); L[i].m2 = GV(true);
    L[i].f0 = GV(true); L[i].f1 = GV(true);
    L[i].l0g = P(); L[i].l0b = P(); L[i].l1g = P(); L[i].l1b = P();
  }
  GvpP rWo; rWo.wh = P(); rWo.ws_w = P(); rWo.ws_b = P();
  rWo.wv = nullptr; rWo.wsv_w = nullptr; rWo.wsv_b = nullptr;
  const float* w1 = P(); const float* b1 = P();
  const float* w2 = P(); const float* b2 = P();

  // --- workspace allocator (256B aligned) ---
  char* wsp = (char*)d_ws;
  auto alloc_h = [&](size_t nh) {
    _Float16* q = (_Float16*)wsp; wsp += ((nh * 2 + 255) / 256) * 256; return q;
  };
  auto alloc_f = [&](size_t nf) {
    float* q = (float*)wsp; wsp += ((nf * 4 + 255) / 256) * 256; return q;
  };
  auto pad16 = [](int x) { return (x + 15) & ~15; };
  auto pad32 = [](int x) { return (x + 31) & ~31; };
  auto prepW = [&](const float* W, int K, int N) -> const _Float16* {
    int KP = pad32(K), NP = pad16(N);
    _Float16* dst = alloc_h((size_t)NP * KP);
    int tot = NP * KP;
    prep_wt_kernel<<<(tot + 255) / 256, 256, 0, stream>>>(W, dst, K, N, KP, NP);
    return dst;
  };
  auto prepB = [&](const float* b, int N) -> const float* {
    int NP = pad16(N);
    float* dst = alloc_f(NP);
    prep_bias_kernel<<<(NP + 255) / 256, 256, 0, stream>>>(b, dst, N, NP);
    return dst;
  };
  auto mkgvp = [&](const GvpP& g, int Kh, int Nh, int Ks, int Ns, int Kv, int Nv,
                   int Kg, int Ng) -> GvpW {
    GvpW w{};
    w.wh = prepW(g.wh, Kh, Nh);
    w.ws = prepW(g.ws_w, Ks, Ns);
    w.bs = prepB(g.ws_b, Ns);
    if (g.wv) w.wv = prepW(g.wv, Kv, Nv);
    if (g.wsv_w) { w.wsv = prepW(g.wsv_w, Kg, Ng); w.bsv = prepB(g.wsv_b, Ng); }
    return w;
  };

  GvpW Wv = mkgvp(rWv, 3, 16, 22, 100, 16, 16, 0, 0);
  GvpW We{};
  We.ws = prepW(rWe.ws_w, 33, 32);
  We.bs = prepB(rWe.ws_b, 32);
  struct LayerW { GvpW m0, m1, m2, f0, f1; } LW[5];
  for (int i = 0; i < 5; ++i) {
    LW[i].m0 = mkgvp(L[i].m0, 33, 33, 265, 100, 33, 16, 100, 16);
    LW[i].m1 = mkgvp(L[i].m1, 16, 16, 116, 100, 16, 16, 100, 16);
    LW[i].m2 = mkgvp(L[i].m2, 16, 16, 116, 100, 16, 16, 100, 16);
    LW[i].f0 = mkgvp(L[i].f0, 16, 32, 132, 400, 32, 32, 400, 32);
    LW[i].f1 = mkgvp(L[i].f1, 32, 32, 432, 100, 32, 16, 100, 16);
  }
  GvpW Wo = mkgvp(rWo, 16, 16, 116, 100, 0, 0, 0, 0);

  float* s      = alloc_f((size_t)NNODES * NSd);
  float* v      = alloc_f((size_t)NNODES * 48);
  float* es     = alloc_f((size_t)NEDGES * 32);
  float* ev     = alloc_f((size_t)NEDGES * 3);
  float* dsum   = alloc_f((size_t)NNODES * NSd);
  float* dvsum  = alloc_f((size_t)NNODES * 48);
  float* cnt    = alloc_f(NNODES);
  float* pooled = alloc_f(NGRAPH * NSd);
  float* gcnt   = alloc_f(NGRAPH);

  const int ZT = 256;
  zero_kernel<<<(NNODES + ZT - 1) / ZT, ZT, 0, stream>>>(cnt, NNODES);
  zero_kernel<<<(NGRAPH * NSd + ZT - 1) / ZT, ZT, 0, stream>>>(pooled, NGRAPH * NSd);
  zero_kernel<<<1, ZT, 0, stream>>>(gcnt, NGRAPH);
  degree_kernel<<<(NEDGES + ZT - 1) / ZT, ZT, 0, stream>>>(ei, cnt);
  gcount_kernel<<<(NNODES + ZT - 1) / ZT, ZT, 0, stream>>>(bat, gcnt);

  const int ntiles = (NNODES + 15) / 16;
  const int etiles = (NEDGES + 15) / 16;
  embed_nodes_kernel<<<ntiles, 128, 0, stream>>>(hVs, hVv, Wv, s, v);
  embed_edges_kernel<<<etiles, 128, 0, stream>>>(hEs, hEv, We, rWe.wh, rWe.wv, es, ev);

  for (int l = 0; l < 5; ++l) {
    zero_kernel<<<(NNODES * NSd + ZT - 1) / ZT, ZT, 0, stream>>>(dsum, NNODES * NSd);
    zero_kernel<<<(NNODES * 48 + ZT - 1) / ZT, ZT, 0, stream>>>(dvsum, NNODES * 48);
    edge_msg_kernel<<<etiles, 128, 0, stream>>>(s, v, es, ev, ei,
                                                LW[l].m0, LW[l].m1, LW[l].m2, dsum, dvsum);
    node_update_kernel<<<ntiles, 128, 0, stream>>>(s, v, dsum, dvsum, cnt,
                                                   LW[l].f0, LW[l].f1,
                                                   L[l].l0g, L[l].l0b, L[l].l1g, L[l].l1b);
  }
  out_pool_kernel<<<ntiles, 128, 0, stream>>>(s, v, bat, Wo, pooled);
  dense_kernel<<<1, 64, 0, stream>>>(pooled, gcnt, w1, b1, w2, b2, (float*)d_out);

  (void)in_sizes; (void)n_in; (void)out_size; (void)ws_size;
}
